// GCN_40699110097466
// MI455X (gfx1250) — matrix operand order repaired and checked by reading, not compiled, not run on hardware
//
#include <hip/hip_runtime.h>
#include <math.h>

#define D 384
#define NTILES_N (D / 16)    // 24
#define NB 4                 // N-blocking: each wave computes 16 x (NB*16) = 16x64
#define NGROUPS (NTILES_N / NB)  // 6

typedef __attribute__((ext_vector_type(16))) __bf16 v16bf;
typedef __attribute__((ext_vector_type(8)))  float  v8f;

union V16 { v16bf v; uint4 q[2]; };

__device__ __forceinline__ unsigned short f2bf(float f) {
    union { float f; unsigned u; } v; v.f = f;
    unsigned r = v.u + 0x7FFFu + ((v.u >> 16) & 1u);   // round-to-nearest-even
    return (unsigned short)(r >> 16);
}

// ---------------- zero fp32 buffer ----------------
__global__ void zero_f32(float* p, int n) {
    int i = blockIdx.x * blockDim.x + threadIdx.x;
    if (i < n) p[i] = 0.0f;
}

// ---------------- weight prep: fp32 [K,N] row-major -> bf16 [N,K] col-major ----------------
__global__ void prep_weights(const float* w0, const float* w1, const float* w2, const float* w3,
                             unsigned short* t0, unsigned short* t1,
                             unsigned short* t2, unsigned short* t3) {
    int id = blockIdx.x * blockDim.x + threadIdx.x;
    if (id >= 4 * D * D) return;
    int mat = id / (D * D);
    int idx = id - mat * (D * D);
    int k = idx / D, n = idx - k * D;
    const float* w = (mat == 0) ? w0 : (mat == 1) ? w1 : (mat == 2) ? w2 : w3;
    unsigned short* t = (mat == 0) ? t0 : (mat == 1) ? t1 : (mat == 2) ? t2 : t3;
    t[(size_t)n * D + k] = f2bf(w[(size_t)k * D + n]);
}

// ---------------- edge scatter: agg[dst] += ew * x[src]; cnt[dst] += 1 ----------------
// one wave per edge; 3 x float4 per lane covers 384 dims; all traffic is L2-resident
__global__ void scatter_edges(const float* __restrict__ X,
                              const int* __restrict__ ei, const float* __restrict__ ew,
                              float* __restrict__ agg, float* __restrict__ cnt, int E) {
    int wave = blockIdx.x * (blockDim.x >> 5) + (threadIdx.x >> 5);
    int lane = threadIdx.x & 31;
    if (wave >= E) return;
    int s = ei[wave];
    int d = ei[E + wave];
    float w = ew[wave];
    if (lane == 0) atomicAdd(&cnt[d], 1.0f);
    const float* xs = X + (size_t)s * D;
    float* ad = agg + (size_t)d * D;
#pragma unroll
    for (int c = 0; c < 3; ++c) {
        int base = c * 128 + lane * 4;
        float4 v = *(const float4*)(xs + base);
        atomicAdd(ad + base + 0, w * v.x);
        atomicAdd(ad + base + 1, w * v.y);
        atomicAdd(ad + base + 2, w * v.z);
        atomicAdd(ad + base + 3, w * v.w);
    }
}

// ---------------- mean = agg / max(cnt,1), converted to bf16 ----------------
__global__ void mean_to_bf16(const float* __restrict__ agg, const float* __restrict__ cnt,
                             unsigned short* __restrict__ out, int n) {
    int i = blockIdx.x * blockDim.x + threadIdx.x;
    if (i >= n) return;
    float c = cnt[i / D];
    c = fmaxf(c, 1.0f);
    out[i] = f2bf(agg[i] / c);
}

// ---------------- fp32 -> bf16 convert ----------------
__global__ void conv_bf16(const float* __restrict__ src, unsigned short* __restrict__ dst, int n) {
    int i = blockIdx.x * blockDim.x + threadIdx.x;
    if (i < n) dst[i] = f2bf(src[i]);
}

// ---------------- fused WMMA GEMM: H = relu(Mean@Wrel + X@Wroot + b) ----------------
// one wave per 16x64 output strip: NB=4 accumulators, A fragment reused across 4 WMMAs.
// bf16 x bf16 -> fp32 accumulate; M,N,K all divide evenly -> EXEC all-ones at every WMMA.
__global__ void gemm_wmma(const unsigned short* __restrict__ Ab,     // mean, bf16 [M,384] row-major
                          const unsigned short* __restrict__ Xb,     // root input, bf16 [M,384]
                          const unsigned short* __restrict__ WrelT,  // bf16 [384,384] col-major
                          const unsigned short* __restrict__ WrootT, // bf16 [384,384] col-major
                          const float* __restrict__ bias,
                          float* __restrict__ Hout) {
    int wave = blockIdx.x * (blockDim.x >> 5) + (threadIdx.x >> 5);
    int ng = wave % NGROUPS;          // which 64-wide column group
    int mt = wave / NGROUPS;          // which 16-row strip
    int lane = threadIdx.x & 31;
    int l = lane & 15;
    int half = lane >> 4;
    int m0 = mt * 16, n0 = ng * (NB * 16);

    v8f acc[NB];
#pragma unroll
    for (int j = 0; j < NB; ++j) {
        float bv = bias[n0 + j * 16 + l];
#pragma unroll
        for (int r = 0; r < 8; ++r) acc[j][r] = bv;
    }

    const unsigned short* Arow = Ab + (size_t)(m0 + l) * D;
    const unsigned short* Xrow = Xb + (size_t)(m0 + l) * D;
    const unsigned short* Wc1[NB];
    const unsigned short* Wc2[NB];
#pragma unroll
    for (int j = 0; j < NB; ++j) {
        Wc1[j] = WrelT  + (size_t)(n0 + j * 16 + l) * D;
        Wc2[j] = WrootT + (size_t)(n0 + j * 16 + l) * D;
    }

    // pass 1: mean @ W_rel
#pragma unroll 2
    for (int k0 = 0; k0 < D; k0 += 32) {
        V16 a;
        a.q[0] = *(const uint4*)(Arow + k0 + half * 8);        // A VGPR0-3: K chunk
        a.q[1] = *(const uint4*)(Arow + k0 + 16 + half * 8);   // A VGPR4-7: K chunk + 16
#pragma unroll
        for (int j = 0; j < NB; ++j) {
            V16 b;
            b.q[0] = *(const uint4*)(Wc1[j] + k0 + half * 8);
            b.q[1] = *(const uint4*)(Wc1[j] + k0 + 16 + half * 8);
            acc[j] = __builtin_amdgcn_wmma_f32_16x16x32_bf16(false, a.v, false, b.v,
                                                             (short)0, acc[j], false, false);
        }
    }
    // pass 2: x @ W_root (accumulates into same tiles)
#pragma unroll 2
    for (int k0 = 0; k0 < D; k0 += 32) {
        V16 a;
        a.q[0] = *(const uint4*)(Xrow + k0 + half * 8);
        a.q[1] = *(const uint4*)(Xrow + k0 + 16 + half * 8);
#pragma unroll
        for (int j = 0; j < NB; ++j) {
            V16 b;
            b.q[0] = *(const uint4*)(Wc2[j] + k0 + half * 8);
            b.q[1] = *(const uint4*)(Wc2[j] + k0 + 16 + half * 8);
            acc[j] = __builtin_amdgcn_wmma_f32_16x16x32_bf16(false, a.v, false, b.v,
                                                             (short)0, acc[j], false, false);
        }
    }

    // epilogue: bias already folded in; ReLU + fp32 store
#pragma unroll
    for (int j = 0; j < NB; ++j) {
#pragma unroll
        for (int r = 0; r < 8; ++r) {
            int row = m0 + r + half * 8;
            float v = acc[j][r];
            Hout[(size_t)row * D + n0 + j * 16 + l] = v > 0.0f ? v : 0.0f;
        }
    }
}

// ---------------- head: out = sigmoid(y @ W_lin + b_lin), OUT=2 ----------------
__global__ void head_kernel(const float* __restrict__ Y, const float* __restrict__ Wl,
                            const float* __restrict__ bl, float* __restrict__ out, int Nn) {
    int node = blockIdx.x * (blockDim.x >> 5) + (threadIdx.x >> 5);
    int lane = threadIdx.x & 31;
    if (node >= Nn) return;
    const float* yr = Y + (size_t)node * D;
    float a0 = 0.0f, a1 = 0.0f;
#pragma unroll
    for (int c = 0; c < D / 32; ++c) {
        int k = c * 32 + lane;
        float v = yr[k];
        a0 += v * Wl[k * 2 + 0];
        a1 += v * Wl[k * 2 + 1];
    }
#pragma unroll
    for (int off = 16; off > 0; off >>= 1) {
        a0 += __shfl_xor(a0, off, 32);
        a1 += __shfl_xor(a1, off, 32);
    }
    if (lane == 0) {
        out[(size_t)node * 2 + 0] = 1.0f / (1.0f + expf(-(a0 + bl[0])));
        out[(size_t)node * 2 + 1] = 1.0f / (1.0f + expf(-(a1 + bl[1])));
    }
}

extern "C" void kernel_launch(void* const* d_in, const int* in_sizes, int n_in,
                              void* d_out, int out_size, void* d_ws, size_t ws_size,
                              hipStream_t stream) {
    const float* x       = (const float*)d_in[0];
    const int*   ei      = (const int*)d_in[1];
    const float* ew      = (const float*)d_in[2];
    const float* W_rel1  = (const float*)d_in[3];
    const float* b_rel1  = (const float*)d_in[4];
    const float* W_root1 = (const float*)d_in[5];
    const float* W_rel2  = (const float*)d_in[6];
    const float* b_rel2  = (const float*)d_in[7];
    const float* W_root2 = (const float*)d_in[8];
    const float* W_lin   = (const float*)d_in[9];
    const float* b_lin   = (const float*)d_in[10];

    const int N = in_sizes[0] / D;   // 100000
    const int E = in_sizes[2];       // 800000
    const size_t ND = (size_t)N * D;

    float* out = (float*)d_out;
    float* y   = out + (size_t)N * 2;   // y region of the tuple output

    // workspace carve-up (256B aligned pieces)
    char* ws = (char*)d_ws;
    size_t o = 0;
    auto carve = [&](size_t bytes) { void* p = ws + o; o = (o + bytes + 255) & ~(size_t)255; return p; };
    float*          agg   = (float*)carve(ND * sizeof(float));
    float*          cnt   = (float*)carve((size_t)N * sizeof(float));
    unsigned short* meanb = (unsigned short*)carve(ND * 2);
    unsigned short* xb    = (unsigned short*)carve(ND * 2);
    float*          h     = (float*)carve(ND * sizeof(float));
    unsigned short* Wr1T  = (unsigned short*)carve((size_t)D * D * 2);
    unsigned short* Wo1T  = (unsigned short*)carve((size_t)D * D * 2);
    unsigned short* Wr2T  = (unsigned short*)carve((size_t)D * D * 2);
    unsigned short* Wo2T  = (unsigned short*)carve((size_t)D * D * 2);

    const int TPB = 256;
    int gemm_blocks = (N / 16) * NGROUPS / 4;         // 4 waves (16x64 strips) per 128-thread block
    int wave_blocks_E = (E + 3) / 4;                  // 4 waves per 128-thread block
    int wave_blocks_N = (N + 3) / 4;
    int nd_blocks = (int)((ND + TPB - 1) / TPB);

    // weight prep (cheap, 4 * 384*384)
    prep_weights<<<(4 * D * D + TPB - 1) / TPB, TPB, 0, stream>>>(
        W_rel1, W_root1, W_rel2, W_root2, Wr1T, Wo1T, Wr2T, Wo2T);

    // x -> bf16 once
    conv_bf16<<<nd_blocks, TPB, 0, stream>>>(x, xb, (int)ND);

    // ---- layer 1 ----
    zero_f32<<<(int)((ND + N + TPB - 1) / TPB), TPB, 0, stream>>>(agg, (int)(ND + N));
    scatter_edges<<<wave_blocks_E, 128, 0, stream>>>(x, ei, ew, agg, cnt, E);
    mean_to_bf16<<<nd_blocks, TPB, 0, stream>>>(agg, cnt, meanb, (int)ND);
    gemm_wmma<<<gemm_blocks, 128, 0, stream>>>(meanb, xb, Wr1T, Wo1T, b_rel1, h);

    // ---- layer 2 ----
    zero_f32<<<(int)((ND + N + TPB - 1) / TPB), TPB, 0, stream>>>(agg, (int)(ND + N));
    scatter_edges<<<wave_blocks_E, 128, 0, stream>>>(h, ei, ew, agg, cnt, E);
    mean_to_bf16<<<nd_blocks, TPB, 0, stream>>>(agg, cnt, meanb, (int)ND);
    conv_bf16<<<nd_blocks, TPB, 0, stream>>>(h, xb, (int)ND);          // bf16(h) for root path
    gemm_wmma<<<gemm_blocks, 128, 0, stream>>>(meanb, xb, Wr2T, Wo2T, b_rel2, y);

    // ---- head ----
    head_kernel<<<wave_blocks_N, 128, 0, stream>>>(y, W_lin, b_lin, out, N);

    (void)n_in; (void)out_size; (void)ws_size; (void)cnt;
}
